// LA_TPP_Model_18468359372882
// MI455X (gfx1250) — matrix operand order repaired and checked by reading, not compiled, not run on hardware
//
#include <hip/hip_runtime.h>

// Problem constants (match reference)
#define B_DIM 2048
#define P_DIM 2048
#define R_DIM 4096
#define H_DIM 6
#define M_DIM 256
#define A_DIM 256
#define OUTW  (P_DIM + M_DIM + A_DIM)   // 2560
#define TAU_F   0.1f
#define BETA_F  5.0f

typedef __attribute__((ext_vector_type(16))) __bf16        v16bf;
typedef __attribute__((ext_vector_type(8)))  float         v8f;
typedef __attribute__((ext_vector_type(8)))  unsigned int  v8u;

__device__ __forceinline__ unsigned f2bf_bits(float x) {
  unsigned u = __float_as_uint(x);
  return (u + 0x7FFFu + ((u >> 16) & 1u)) >> 16;   // round-to-nearest-even
}

// CDNA5 async copy engine: memory -> LDS, tracked by ASYNCcnt, no VGPR data.
__device__ __forceinline__ void async_copy_b128(unsigned lds_off,
                                                const unsigned short* gsrc) {
  asm volatile("global_load_async_to_lds_b128 %0, %1, off"
               :: "v"(lds_off), "v"((unsigned long long)(size_t)gsrc)
               : "memory");
}
__device__ __forceinline__ void wait_async0() {
  asm volatile("s_wait_asynccnt 0x0" ::: "memory");
}

// ---------------------------------------------------------------------------
// K1: val[b][p] = max(p<256 ? clip(M_minus,0,1) : 0, action_mask); also Mclip bf16
// ---------------------------------------------------------------------------
__global__ __launch_bounds__(256) void k_prep_val(
    const float* __restrict__ M_minus, const float* __restrict__ action_mask,
    float* __restrict__ val, unsigned short* __restrict__ mclip_bf) {
  int idx = blockIdx.x * 256 + threadIdx.x;        // over B*P
  int b = idx >> 11, p = idx & (P_DIM - 1);
  float am = action_mask[idx];
  float v = am;
  if (p < M_DIM) {
    float mc = fminf(fmaxf(M_minus[b * M_DIM + p], 0.f), 1.f);
    v = fmaxf(mc, am);
    mclip_bf[b * M_DIM + p] = (unsigned short)f2bf_bits(mc);
  }
  val[idx] = v;
}

// ---------------------------------------------------------------------------
// K2: Bcomb in tile-contiguous TRANSPOSED layout: tiles of [128 n][32 k],
// tile order (ktile*NT + ntile), NT = 4.  Written in output order (coalesced).
// Bcomb(k,n) = n<256 ? is_toM[k]*Gamma_AtoM[k][n] : is_toA[k]*Gamma_MtoA[k][n-256]
// ---------------------------------------------------------------------------
__global__ __launch_bounds__(256) void k_prep_bcomb(
    const float* __restrict__ GammaM, const float* __restrict__ GammaA,
    const float* __restrict__ is_toM, const float* __restrict__ is_toA,
    unsigned short* __restrict__ bcomb) {
  int idx = blockIdx.x * 256 + threadIdx.x;        // over R*512 (output order)
  int kin = idx & 31;
  int nin = (idx >> 5) & 127;
  int t   = idx >> 12;
  int ntile = t & 3, ktile = t >> 2;
  int k = ktile * 32 + kin;
  int n = ntile * 128 + nin;
  float v = (n < 256) ? is_toM[k] * GammaM[k * 256 + n]
                      : is_toA[k] * GammaA[k * 256 + (n - 256)];
  bcomb[idx] = (unsigned short)f2bf_bits(v);
}

// ---------------------------------------------------------------------------
// K3: wT in the same tiled layout, NT = 2.  wT(k=m, n=a) = w[a][m].
// ---------------------------------------------------------------------------
__global__ __launch_bounds__(256) void k_prep_wT(
    const float* __restrict__ w, unsigned short* __restrict__ wT) {
  int idx = blockIdx.x * 256 + threadIdx.x;        // over 256*256 (output order)
  int kin = idx & 31;
  int nin = (idx >> 5) & 127;
  int t   = idx >> 12;
  int ntile = t & 1, ktile = t >> 1;
  int k = ktile * 32 + kin;
  int n = ntile * 128 + nin;
  wT[idx] = (unsigned short)f2bf_bits(w[n * M_DIM + k]);
}

// ---------------------------------------------------------------------------
// K4: g[b][r] = gate * ( -TAU * logsumexp(-v/TAU) );  write fp32 (for seg-LSE)
// and bf16 (GEMM A operand).
// ---------------------------------------------------------------------------
__global__ __launch_bounds__(256) void k_gbody(
    const float* __restrict__ val, const float* __restrict__ gate,
    const int* __restrict__ body_idx, const float* __restrict__ body_mask,
    float* __restrict__ g, unsigned short* __restrict__ gbf) {
  int idx = blockIdx.x * 256 + threadIdx.x;        // over B*R
  int b = idx >> 12, r = idx & (R_DIM - 1);
  const float* vrow = val + (size_t)b * P_DIM;
  float v[H_DIM];
  float mn = 1e30f;
#pragma unroll
  for (int h = 0; h < H_DIM; ++h) {
    int p = body_idx[r * H_DIM + h];
    float bm = body_mask[r * H_DIM + h];
    float bv = vrow[p];
    v[h] = (bm > 0.f) ? bv : 10000.f;
    mn = fminf(mn, v[h]);
  }
  float s = 0.f;
#pragma unroll
  for (int h = 0; h < H_DIM; ++h) s += __expf((mn - v[h]) * (1.0f / TAU_F));
  float gv = (mn - TAU_F * __logf(s)) * gate[idx];
  g[idx] = gv;
  gbf[idx] = (unsigned short)f2bf_bits(gv);
}

// ---------------------------------------------------------------------------
// K5: per batch-row segment-LSE over head_idx via LDS atomics; write val_new
// ---------------------------------------------------------------------------
__global__ __launch_bounds__(256) void k_seg_val(
    const float* __restrict__ g, const float* __restrict__ val,
    const int* __restrict__ head_idx, float* __restrict__ out) {
  __shared__ unsigned mxu[P_DIM];   // order-preserving float encoding
  __shared__ float    sm[P_DIM];
  const int b = blockIdx.x, tid = threadIdx.x;
  for (int p = tid; p < P_DIM; p += 256) { mxu[p] = 0u; sm[p] = 0.f; }
  __syncthreads();
  const float* grow = g + (size_t)b * R_DIM;
  for (int r = tid; r < R_DIM; r += 256) {
    float gv = BETA_F * grow[r];
    unsigned u = __float_as_uint(gv);
    unsigned enc = (u & 0x80000000u) ? ~u : (u | 0x80000000u);
    atomicMax(&mxu[head_idx[r]], enc);
  }
  __syncthreads();
  for (int r = tid; r < R_DIM; r += 256) {
    int h = head_idx[r];
    float gv = BETA_F * grow[r];
    unsigned e = mxu[h];
    float m = (e & 0x80000000u) ? __uint_as_float(e & 0x7FFFFFFFu)
                                : __uint_as_float(~e);
    float msafe = (m > -1e30f) ? m : 0.f;
    atomicAdd(&sm[h], __expf(gv - msafe));
  }
  __syncthreads();
  const float* vrow = val + (size_t)b * P_DIM;
  float* orow = out + (size_t)b * OUTW;
  for (int p = tid; p < P_DIM; p += 256) {
    float v = vrow[p];
    float o = v;
    if (p < M_DIM) {
      float s = sm[p];
      if (s > 0.f) {
        unsigned e = mxu[p];
        float m = (e & 0x80000000u) ? __uint_as_float(e & 0x7FFFFFFFu)
                                    : __uint_as_float(~e);
        float msafe = (m > -1e30f) ? m : 0.f;
        float lse = (msafe + __logf(fmaxf(s, 1e-30f))) * (1.0f / BETA_F);
        o = fmaxf(v, lse);
      }  // s==0 -> lse = -1e9 -> max is v
    }
    orow[p] = o;
  }
}

// ---------------------------------------------------------------------------
// K6/K7: bf16 WMMA GEMM, tile 64x128x32, 8 waves (wave32), 16x64 per wave.
// Double-buffered LDS staged entirely with global_load_async_to_lds_b128.
// A: bf16 row-major.  B: bf16 tiled [128n x 32k] per (ktile,ntile) tile.
// MODE 0: out[:,2304:2560] = softplus(bias + A@B)            (lam_ment)
// MODE 1: cols<256 -> out[:,2048:2304] = relu(A@B)           (S)
//         cols>=256 -> out[:,2304:2560] += relu(A@B)         (lam_logic)
// ---------------------------------------------------------------------------
#define BM 64
#define BN 128
#define BK 32
#define LDSP 40   // row stride in shorts: 80 B (16 B multiple -> aligned b128)

template<int MODE>
__global__ __launch_bounds__(256) void k_gemm_wmma(
    const unsigned short* __restrict__ Abf,
    const unsigned short* __restrict__ Btiled,
    const float* __restrict__ bias, float* __restrict__ out,
    int K, int lda, int NT) {
  __shared__ unsigned short As[2][BM][LDSP];
  __shared__ unsigned short Bs[2][BN][LDSP];
  const int tid = threadIdx.x;
  const int wave = tid >> 5, lane = tid & 31;
  const int wy = wave >> 1, wx = wave & 1;         // 4x2 wave grid
  const int halfL = lane >> 4, l16 = lane & 15;
  const int blockRow = blockIdx.x * BM;
  const int blockCol = blockIdx.y * BN;

  v8f acc[4];
  const v8f vzero = {0.f, 0.f, 0.f, 0.f, 0.f, 0.f, 0.f, 0.f};
#pragma unroll
  for (int i = 0; i < 4; ++i) acc[i] = vzero;

  // staging map: A: 1 chunk/thread; B: 2 chunks/thread (3 async ops / wave / tile)
  const int arow = tid >> 2, aq = tid & 3;         // A: 64 rows x 4 chunks
  const int brow = tid >> 1, bh = tid & 1;         // B: 128 rows x 2 half-rows

  const unsigned short* aBase = Abf + (size_t)(blockRow + arow) * lda + aq * 8;
  const unsigned aLds[2] = {(unsigned)(size_t)&As[0][arow][aq * 8],
                            (unsigned)(size_t)&As[1][arow][aq * 8]};
  const unsigned bLds[2] = {(unsigned)(size_t)&Bs[0][brow][bh * 16],
                            (unsigned)(size_t)&Bs[1][brow][bh * 16]};
  const size_t bInTile = (size_t)brow * BK + bh * 16;
  const int ntile = blockCol >> 7;

  // stage tile kt=0 into buffer 0
  {
    const unsigned short* bt = Btiled + (size_t)ntile * (BN * BK) + bInTile;
    async_copy_b128(aLds[0], aBase);
    async_copy_b128(bLds[0], bt);
    async_copy_b128(bLds[0] + 16, bt + 8);
  }

  for (int kt = 0; kt < K; kt += BK) {
    const int cur = (kt >> 5) & 1;
    wait_async0();        // this wave's copies for tile kt are in LDS
    __syncthreads();      // all waves' copies done; prev reads of buf cur^1 done
    if (kt + BK < K) {    // prefetch next tile into the other buffer (async)
      const int nb = cur ^ 1;
      const unsigned short* bt =
          Btiled + ((size_t)((kt + BK) >> 5) * NT + ntile) * (BN * BK) + bInTile;
      async_copy_b128(aLds[nb], aBase + kt + BK);
      async_copy_b128(bLds[nb], bt);
      async_copy_b128(bLds[nb] + 16, bt + 8);
    }

    // ---- A fragment: 16-bit A 16x32 (lanes 0-15: K 0..7|16..23; 16-31: 8..15|24..31)
    v8u au;
    {
      const int row = wy * 16 + l16;
#pragma unroll
      for (int j = 0; j < 8; ++j) {
        const int kb = ((j < 4) ? 0 : 16) + halfL * 8 + 2 * (j & 3);
        au[j] = *(const unsigned*)&As[cur][row][kb];
      }
    }
    const v16bf afrag = __builtin_bit_cast(v16bf, au);

#pragma unroll
    for (int cx = 0; cx < 4; ++cx) {
      // ---- B fragment: 32x16, lanes 0-15: K 0..7|16..23, lanes 16-31: K 8..15|24..31 (pairs)
      v8u bu;
      const int col = wx * 64 + cx * 16 + l16;
#pragma unroll
      for (int j = 0; j < 8; ++j) {
        const int kb = ((j < 4) ? 0 : 16) + halfL * 8 + 2 * (j & 3);
        bu[j] = *(const unsigned*)&Bs[cur][col][kb];
      }
      const v16bf bfrag = __builtin_bit_cast(v16bf, bu);
      acc[cx] = __builtin_amdgcn_wmma_f32_16x16x32_bf16(
          false, afrag, false, bfrag, (short)0, acc[cx], false, false);
    }
  }

  // ---- epilogue (C layout: lane n = l16, row m = vgpr + 8*halfL)
#pragma unroll
  for (int cx = 0; cx < 4; ++cx) {
#pragma unroll
    for (int v = 0; v < 8; ++v) {
      const int gr = blockRow + wy * 16 + v + 8 * halfL;
      const int gc = blockCol + wx * 64 + cx * 16 + l16;
      const float a = acc[cx][v];
      if (MODE == 0) {
        float x = bias[gc] + a;
        float sp = (x > 20.f) ? x : __logf(1.0f + __expf(x));
        out[(size_t)gr * OUTW + (P_DIM + M_DIM) + gc] = sp;
      } else {
        float rl = fmaxf(a, 0.f);
        if (gc < 256) {
          out[(size_t)gr * OUTW + P_DIM + gc] = rl;
        } else {
          size_t o = (size_t)gr * OUTW + (P_DIM + M_DIM) + (gc - 256);
          out[o] += rl;   // ordered after MODE 0 kernel on same stream
        }
      }
    }
  }
}

// ---------------------------------------------------------------------------
extern "C" void kernel_launch(void* const* d_in, const int* in_sizes, int n_in,
                              void* d_out, int out_size, void* d_ws, size_t ws_size,
                              hipStream_t stream) {
  const float* M_minus     = (const float*)d_in[0];
  const float* action_mask = (const float*)d_in[1];
  const float* gate        = (const float*)d_in[2];
  const float* body_mask   = (const float*)d_in[3];
  const float* is_toM      = (const float*)d_in[4];
  const float* is_toA      = (const float*)d_in[5];
  const float* Gamma_AtoM  = (const float*)d_in[6];
  const float* Gamma_MtoA  = (const float*)d_in[7];
  const float* bvec        = (const float*)d_in[8];
  const float* w           = (const float*)d_in[9];
  const int*   body_idx    = (const int*)d_in[10];
  const int*   head_idx    = (const int*)d_in[11];
  // d_in[12] mental_pos == arange(256): identity mapping assumed.
  float* out = (float*)d_out;

  // Workspace layout (~69 MB total)
  char* ws = (char*)d_ws;
  float* ws_val = (float*)ws;                                   // B*P f32 (16MB)
  float* ws_g   = (float*)(ws + (size_t)16 * 1024 * 1024);      // B*R f32 (32MB)
  unsigned short* ws_gbf =
      (unsigned short*)(ws + (size_t)48 * 1024 * 1024);         // B*R bf16 (16MB)
  unsigned short* ws_bcomb =
      (unsigned short*)(ws + (size_t)64 * 1024 * 1024);         // R*512 bf16 (4MB)
  unsigned short* ws_mclip =
      (unsigned short*)(ws + (size_t)68 * 1024 * 1024);         // B*M bf16 (1MB)
  unsigned short* ws_wT =
      (unsigned short*)(ws + (size_t)69 * 1024 * 1024);         // M*A bf16 (128KB)

  k_prep_val  <<<B_DIM * P_DIM / 256, 256, 0, stream>>>(M_minus, action_mask, ws_val, ws_mclip);
  k_prep_bcomb<<<R_DIM * 512 / 256, 256, 0, stream>>>(Gamma_AtoM, Gamma_MtoA, is_toM, is_toA, ws_bcomb);
  k_prep_wT   <<<M_DIM * A_DIM / 256, 256, 0, stream>>>(w, ws_wT);
  k_gbody     <<<B_DIM * R_DIM / 256, 256, 0, stream>>>(ws_val, gate, body_idx, body_mask, ws_g, ws_gbf);
  k_seg_val   <<<B_DIM, 256, 0, stream>>>(ws_g, ws_val, head_idx, out);

  // lam_ment = softplus(b + Mclip @ w^T)  -> lam region (runs before main GEMM)
  dim3 g0(B_DIM / BM, A_DIM / BN);   // (32, 2)
  k_gemm_wmma<0><<<g0, 256, 0, stream>>>(ws_mclip, ws_wT, bvec, out,
                                         M_DIM, M_DIM, 2);
  // [S | lam_logic] = g @ Bcomb   (K=4096, N=512)
  dim3 g1(B_DIM / BM, 512 / BN);     // (32, 4)
  k_gemm_wmma<1><<<g1, 256, 0, stream>>>(ws_gbf, ws_bcomb, nullptr, out,
                                         R_DIM, R_DIM, 4);
}
